// VGD_GNN_65798898975054
// MI455X (gfx1250) — matrix operand order repaired and checked by reading, not compiled, not run on hardware
//
#include <hip/hip_runtime.h>
#include <hip/hip_bf16.h>
#include <math.h>

typedef __attribute__((ext_vector_type(16))) _Float16 v16h;
typedef __attribute__((ext_vector_type(8)))  float    v8f;

#define WAVE 32

// ---------------------------------------------------------------------------
// WMMA GEMM:  C[M,N] = act( A[M,K] * B[K,N] + bias )
// One wave32 computes a 32x32 block of C as a 2x2 grid of 16x16 WMMA tiles,
// looping K in steps of 32 with v_wmma_f32_16x16x32_f16 (f32 accumulate).
// 4 independent WMMAs per K-step amortize fragment loads 2x in each direction
// while staying spill-free (acc 32 + A 16 + B 16 VGPRs core state).
// A fragments are loaded as float4 (16B-aligned by construction: 256B-aligned
// base, row stride K%32==0, k offsets %8 floats) so they lower to
// global_load_b128; B's column-major per-lane walk stays b32 but is fully
// line-efficient (lanes 0-15 / 16-31 each cover whole 64B lines).
// Requires M%32==0, N%32==0, K%32==0 (true for every GEMM in this model).
// Operand packing follows CDNA5 ISA 7.12.2 VGPR layouts.
// ---------------------------------------------------------------------------
__global__ __launch_bounds__(256, 1)
void gemm_wmma_k(const float* __restrict__ A, const float* __restrict__ B,
                 const float* __restrict__ bias, float* __restrict__ C,
                 int M, int N, int K, int relu)
{
    const int wavesPerBlock = blockDim.x / WAVE;
    const int wave = blockIdx.x * wavesPerBlock + ((int)threadIdx.x / WAVE);
    const int lane = threadIdx.x & (WAVE - 1);
    const int tilesN = N >> 5;                 // 32-wide wave blocks
    const int tiles  = (M >> 5) * tilesN;      // 32-tall wave blocks
    if (wave >= tiles) return;                 // wave-uniform: EXEC all-ones inside
    const int row0 = (wave / tilesN) << 5;
    const int col0 = (wave % tilesN) << 5;
    const int m    = lane & 15;
    const int half = lane >> 4;

    v8f acc[2][2] = {};
#pragma unroll 1
    for (int k = 0; k < K; k += 32) {
        v16h a[2], b[2];
        // A fragments: lane holds row (row0+r*16+m); elems 0..7 -> K=k+half*8+j,
        // elems 8..15 -> K=k+half*8+16+j  (ISA 16-bit A 16x32 layout).
        // float4 loads -> global_load_b128 (addresses 16B aligned).
#pragma unroll
        for (int r = 0; r < 2; ++r) {
            const float4* ap = (const float4*)(A + (size_t)(row0 + r * 16 + m) * K
                                                 + k + half * 8);
            float4 q0 = ap[0], q1 = ap[1];   // K = k+half*8 + 0..7
            float4 q2 = ap[4], q3 = ap[5];   // K = k+half*8 + 16..23
            a[r][0]  = (_Float16)q0.x; a[r][1]  = (_Float16)q0.y;
            a[r][2]  = (_Float16)q0.z; a[r][3]  = (_Float16)q0.w;
            a[r][4]  = (_Float16)q1.x; a[r][5]  = (_Float16)q1.y;
            a[r][6]  = (_Float16)q1.z; a[r][7]  = (_Float16)q1.w;
            a[r][8]  = (_Float16)q2.x; a[r][9]  = (_Float16)q2.y;
            a[r][10] = (_Float16)q2.z; a[r][11] = (_Float16)q2.w;
            a[r][12] = (_Float16)q3.x; a[r][13] = (_Float16)q3.y;
            a[r][14] = (_Float16)q3.z; a[r][15] = (_Float16)q3.w;
        }
        // B fragments: lane holds col (col0+c*16+m); elem j -> K=k+half*8+j (j<8), k+half*8+16+(j-8) (j>=8)
#pragma unroll
        for (int c = 0; c < 2; ++c) {
            const float* bcol = B + (size_t)(k + half * 8) * N + (col0 + c * 16 + m);
#pragma unroll
            for (int j = 0; j < 16; ++j) b[c][j] = (_Float16)bcol[(size_t)(j < 8 ? j : j + 8) * N];
        }
        // prefetch next strided B slice (global_prefetch_b8) to cover L2 latency
        if (k + 32 < K)
            __builtin_prefetch(B + (size_t)(k + 32 + half * 16) * N + col0 + m, 0, 3);
#pragma unroll
        for (int r = 0; r < 2; ++r)
#pragma unroll
            for (int c = 0; c < 2; ++c)
                acc[r][c] = __builtin_amdgcn_wmma_f32_16x16x32_f16(
                    false, a[r], false, b[c], (short)0, acc[r][c], false, false);
    }
    // D layout: VGPR v -> row = row0 + r*16 + half*8 + v, col = col0 + c*16 + m
#pragma unroll
    for (int r = 0; r < 2; ++r) {
#pragma unroll
        for (int c = 0; c < 2; ++c) {
            const int nn = col0 + c * 16 + m;
            const float bv = bias ? bias[nn] : 0.0f;
#pragma unroll
            for (int v = 0; v < 8; ++v) {
                const int mm = row0 + r * 16 + half * 8 + v;
                float val = acc[r][c][v] + bv;
                if (relu) val = fmaxf(val, 0.0f);
                C[(size_t)mm * N + nn] = val;
            }
        }
    }
}

// ---------------------------------------------------------------------------
// fills
// ---------------------------------------------------------------------------
__global__ void fill_f32_k(float* __restrict__ p, float v, long long n)
{
    long long t = (long long)blockIdx.x * blockDim.x + threadIdx.x;
    if (t < n) p[t] = v;
}
__global__ void fill_i32_k(int* __restrict__ p, int v, long long n)
{
    long long t = (long long)blockIdx.x * blockDim.x + threadIdx.x;
    if (t < n) p[t] = v;
}

// deg[dst] += ew  (deg pre-filled with 1.0 for the self loop)
__global__ void degree_k(const int* __restrict__ dst, const float* __restrict__ ew,
                         float* __restrict__ deg, int E)
{
    int e = blockIdx.x * blockDim.x + threadIdx.x;
    if (e < E) atomicAdd(&deg[dst[e]], ew[e]);
}

// agg[dst,:] += rsqrt(deg[src])*rsqrt(deg[dst])*ew * h[src,:]
// one thread handles 4 consecutive features of one edge
__global__ void scatter_k(const int* __restrict__ src, const int* __restrict__ dst,
                          const float* __restrict__ ew, const float* __restrict__ deg,
                          const float* __restrict__ h, float* __restrict__ agg,
                          int E, int H)
{
    long long t = (long long)blockIdx.x * blockDim.x + threadIdx.x;
    int chunks = H >> 2;
    long long e = t / chunks;
    if (e >= E) return;
    int c = (int)(t % chunks) << 2;
    float wgt = ew[e];
    if (wgt == 0.0f) return;                  // pruned edge
    int s = src[e], d = dst[e];
    float nrm = rsqrtf(deg[s]) * rsqrtf(deg[d]) * wgt;
    const float* hs = h + (size_t)s * H + c;
    float* ad = agg + (size_t)d * H + c;
#pragma unroll
    for (int j = 0; j < 4; ++j) atomicAdd(&ad[j], nrm * hs[j]);
}

// act = relu(agg + h/deg + bias)   (act may alias agg: pure elementwise)
__global__ void finalize_k(const float* __restrict__ agg, const float* __restrict__ h,
                           const float* __restrict__ deg, const float* __restrict__ bias,
                           float* __restrict__ act, int n, int H)
{
    long long t = (long long)blockIdx.x * blockDim.x + threadIdx.x;
    if (t >= (long long)n * H) return;
    int i = (int)(t / H), f = (int)(t % H);
    float v = agg[t] + h[t] / deg[i] + bias[f];
    act[t] = fmaxf(v, 0.0f);
}

// ||w|| for TopKPooling (H==256, one block of 256 threads)
__global__ void wnorm_k(const float* __restrict__ w, int H, float* __restrict__ out)
{
    __shared__ float red[256];
    int t = threadIdx.x;
    float v = (t < H) ? w[t] * w[t] : 0.0f;
    red[t] = v;
    __syncthreads();
    for (int s = 128; s > 0; s >>= 1) {
        if (t < s) red[t] += red[t + s];
        __syncthreads();
    }
    if (t == 0) out[0] = sqrtf(red[0]);
}

// Per-graph top-k: block g, npg threads (power of two <= 256).
// score = tanh(x.w/||w||); LDS bitonic sort of (-score, idx); selection set &
// scales match the reference (readout/GCN are order-invariant within a graph).
__global__ void topk_k(const float* __restrict__ act, const float* __restrict__ w,
                       const float* __restrict__ wnorm,
                       int* __restrict__ perm, float* __restrict__ topsc,
                       int* __restrict__ new_id, int npg, int k, int H)
{
    __shared__ float s[256];
    __shared__ int   id[256];
    const int g = blockIdx.x;
    const int i = threadIdx.x;
    const float inv = 1.0f / wnorm[0];
    const float* row = act + ((size_t)g * npg + i) * H;
    float dot = 0.0f;
    for (int f = 0; f < H; ++f) dot += row[f] * w[f];
    s[i]  = -tanhf(dot * inv);   // ascending sort of -score == descending score
    id[i] = i;
    __syncthreads();
    for (int size = 2; size <= npg; size <<= 1) {
        for (int stride = size >> 1; stride > 0; stride >>= 1) {
            int j = i ^ stride;
            if (j > i) {
                bool up = ((i & size) == 0);
                bool sw = up ? (s[i] > s[j]) : (s[i] < s[j]);
                if (sw) {
                    float ts = s[i]; s[i] = s[j]; s[j] = ts;
                    int   ti = id[i]; id[i] = id[j]; id[j] = ti;
                }
            }
            __syncthreads();
        }
    }
    if (i < k) {
        int sel = id[i];
        perm[g * k + i]  = g * npg + sel;
        topsc[g * k + i] = -s[i];
        new_id[g * npg + sel] = g * k + i;
    }
}

// xnew[i,:] = act[perm[i],:] * topsc[i]
__global__ void gather_k(const float* __restrict__ act, const int* __restrict__ perm,
                         const float* __restrict__ sc, float* __restrict__ xnew,
                         long long total, int H)
{
    long long t = (long long)blockIdx.x * blockDim.x + threadIdx.x;
    if (t >= total) return;
    int i = (int)(t / H), f = (int)(t % H);
    xnew[t] = act[(size_t)perm[i] * H + f] * sc[i];
}

// Remap edges through new_id; pruned edges keep slot with ew=0 (as reference)
__global__ void remap_k(const int* __restrict__ src, const int* __restrict__ dst,
                        const float* __restrict__ ew, const int* __restrict__ new_id,
                        int* __restrict__ nsrc, int* __restrict__ ndst,
                        float* __restrict__ new_ew, int E)
{
    int e = blockIdx.x * blockDim.x + threadIdx.x;
    if (e >= E) return;
    int s = new_id[src[e]], d = new_id[dst[e]];
    bool valid = (s >= 0) && (d >= 0);
    new_ew[e] = valid ? ew[e] : 0.0f;
    nsrc[e] = valid ? s : 0;
    ndst[e] = valid ? d : 0;
}

// out[g, 0:H]=segment_max, out[g, H:2H]=segment_mean; batch is contiguous k-blocks
__global__ void readout_k(const float* __restrict__ x, float* __restrict__ out,
                          int k, int H, int accumulate)
{
    int g = blockIdx.x, f = threadIdx.x;   // blockDim.x == H
    float mx = -INFINITY, sum = 0.0f;
    const float* p = x + (size_t)g * k * H + f;
    for (int i = 0; i < k; ++i) {
        float v = p[(size_t)i * H];
        mx = fmaxf(mx, v);
        sum += v;
    }
    float mean = sum / (float)k;
    float* o = out + (size_t)g * 2 * H;
    if (accumulate) { o[f] += mx; o[H + f] += mean; }
    else            { o[f]  = mx; o[H + f]  = mean; }
}

// logits = h2 @ lin3_w + lin3_b ; log_softmax over 2 classes
__global__ void head_k(const float* __restrict__ h2, const float* __restrict__ w,
                       const float* __restrict__ b, float* __restrict__ out,
                       int G, int Kh)
{
    int g = blockIdx.x * blockDim.x + threadIdx.x;
    if (g >= G) return;
    float l0 = b[0], l1 = b[1];
    const float* r = h2 + (size_t)g * Kh;
    for (int j = 0; j < Kh; ++j) { l0 += r[j] * w[2 * j]; l1 += r[j] * w[2 * j + 1]; }
    float mx  = fmaxf(l0, l1);
    float lse = mx + logf(expf(l0 - mx) + expf(l1 - mx));
    out[2 * g]     = l0 - lse;
    out[2 * g + 1] = l1 - lse;
}

// ---------------------------------------------------------------------------
extern "C" void kernel_launch(void* const* d_in, const int* in_sizes, int n_in,
                              void* d_out, int out_size, void* d_ws, size_t ws_size,
                              hipStream_t stream)
{
    (void)n_in; (void)out_size; (void)ws_size;
    const int F_IN = 128, H = 256, G = 256;
    const int n0   = in_sizes[0] / F_IN;   // 65536 nodes
    const int E    = in_sizes[1] / 2;      // 524288 edges
    const int npg0 = n0 / G;               // 256 nodes/graph

    const float* x_in = (const float*)d_in[0];
    const int*   ei   = (const int*)d_in[1];
    const int*   src0 = ei;
    const int*   dst0 = ei + E;
    const float* W0  = (const float*)d_in[3];  const float* b0  = (const float*)d_in[4];
    const float* W1  = (const float*)d_in[5];  const float* b1  = (const float*)d_in[6];
    const float* W2  = (const float*)d_in[7];  const float* b2  = (const float*)d_in[8];
    const float* pw0 = (const float*)d_in[9];
    const float* pw1 = (const float*)d_in[10];
    const float* pw2 = (const float*)d_in[11];
    const float* l1w = (const float*)d_in[12]; const float* l1b = (const float*)d_in[13];
    const float* l2w = (const float*)d_in[14]; const float* l2b = (const float*)d_in[15];
    const float* l3w = (const float*)d_in[16]; const float* l3b = (const float*)d_in[17];

    // bump allocator over d_ws (~180 MB total)
    char* base = (char*)d_ws; size_t off = 0;
    auto alloc = [&](size_t bytes) -> void* {
        void* p = base + off;
        off = (off + bytes + 255) & ~(size_t)255;
        return p;
    };
    float* hbuf = (float*)alloc((size_t)n0 * H * 4);        // pre-agg h = x@W
    float* agg  = (float*)alloc((size_t)n0 * H * 4);        // agg, then act in-place
    float* xcur = (float*)alloc((size_t)(n0 / 2) * H * 4);  // pooled features
    float* deg  = (float*)alloc((size_t)n0 * 4);
    float* ew0  = (float*)alloc((size_t)E * 4);
    int*   srcA = (int*)alloc((size_t)E * 4);
    int*   dstA = (int*)alloc((size_t)E * 4);
    float* ewA  = (float*)alloc((size_t)E * 4);
    int*   srcB = (int*)alloc((size_t)E * 4);
    int*   dstB = (int*)alloc((size_t)E * 4);
    float* ewB  = (float*)alloc((size_t)E * 4);
    int*   perm = (int*)alloc((size_t)G * (npg0 / 2) * 4);
    float* tsc  = (float*)alloc((size_t)G * (npg0 / 2) * 4);
    int*   nid  = (int*)alloc((size_t)n0 * 4);
    float* outa = (float*)alloc((size_t)G * 2 * H * 4);     // readout accumulator
    float* m1   = (float*)alloc((size_t)G * H * 4);
    float* m2   = (float*)alloc((size_t)G * (H / 2) * 4);
    float* wn   = (float*)alloc(256);

    auto fillf = [&](float* p, float v, long long n) {
        fill_f32_k<<<(int)((n + 255) / 256), 256, 0, stream>>>(p, v, n);
    };
    auto filli = [&](int* p, int v, long long n) {
        fill_i32_k<<<(int)((n + 255) / 256), 256, 0, stream>>>(p, v, n);
    };
    auto gemm = [&](const float* A, const float* B, const float* bias, float* C,
                    int M, int N, int K, int relu) {
        int tiles = (M / 32) * (N / 32);              // one wave per 32x32 block
        int blocks = (tiles + 7) / 8;                 // 8 waves / 256-thread block
        gemm_wmma_k<<<blocks, 256, 0, stream>>>(A, B, bias, C, M, N, K, relu);
    };
    auto gcn = [&](const float* X, const float* W, const float* b,
                   const int* src, const int* dst, const float* ew, int n, int K) {
        gemm(X, W, nullptr, hbuf, n, H, K, 0);
        fillf(deg, 1.0f, n);
        degree_k<<<(E + 255) / 256, 256, 0, stream>>>(dst, ew, deg, E);
        fillf(agg, 0.0f, (long long)n * H);
        long long st = (long long)E * (H / 4);
        scatter_k<<<(int)((st + 255) / 256), 256, 0, stream>>>(src, dst, ew, deg,
                                                               hbuf, agg, E, H);
        long long ft = (long long)n * H;
        finalize_k<<<(int)((ft + 255) / 256), 256, 0, stream>>>(agg, hbuf, deg, b,
                                                                agg, n, H);
    };
    auto pool = [&](const float* pw, int n, int npg, int k) {
        wnorm_k<<<1, 256, 0, stream>>>(pw, H, wn);
        filli(nid, -1, n);
        topk_k<<<G, npg, 0, stream>>>(agg, pw, wn, perm, tsc, nid, npg, k, H);
        long long gt = (long long)G * k * H;
        gather_k<<<(int)((gt + 255) / 256), 256, 0, stream>>>(agg, perm, tsc, xcur,
                                                              gt, H);
    };

    fillf(ew0, 1.0f, E);

    // ---- level 0: GCN(128->256) + pool(k=128) ----
    gcn(x_in, W0, b0, src0, dst0, ew0, n0, F_IN);
    pool(pw0, n0, npg0, npg0 / 2);
    remap_k<<<(E + 255) / 256, 256, 0, stream>>>(src0, dst0, ew0, nid,
                                                 srcA, dstA, ewA, E);
    readout_k<<<G, H, 0, stream>>>(xcur, outa, npg0 / 2, H, 0);

    // ---- level 1: GCN(256->256) + pool(k=64) ----
    const int n1 = G * (npg0 / 2), npg1 = npg0 / 2;
    gcn(xcur, W1, b1, srcA, dstA, ewA, n1, H);
    pool(pw1, n1, npg1, npg1 / 2);
    remap_k<<<(E + 255) / 256, 256, 0, stream>>>(srcA, dstA, ewA, nid,
                                                 srcB, dstB, ewB, E);
    readout_k<<<G, H, 0, stream>>>(xcur, outa, npg1 / 2, H, 1);

    // ---- level 2: GCN(256->256) + pool(k=32); edges unused afterwards ----
    const int n2 = G * (npg1 / 2), npg2 = npg1 / 2;
    gcn(xcur, W2, b2, srcB, dstB, ewB, n2, H);
    pool(pw2, n2, npg2, npg2 / 2);
    readout_k<<<G, H, 0, stream>>>(xcur, outa, npg2 / 2, H, 1);

    // ---- MLP head (WMMA GEMMs + fused bias/ReLU), then log_softmax ----
    gemm(outa, l1w, l1b, m1, G, H, 2 * H, 1);       // 256x512 @ 512x256
    gemm(m1,  l2w, l2b, m2, G, H / 2, H, 1);        // 256x256 @ 256x128
    head_k<<<1, 256, 0, stream>>>(m2, l3w, l3b, (float*)d_out, G, H / 2);
}
